// SpikingConvBlock_13821204758975
// MI455X (gfx1250) — matrix operand order repaired and checked by reading, not compiled, not run on hardware
//
#include <hip/hip_runtime.h>

typedef __attribute__((ext_vector_type(16))) __bf16 v16bf;
typedef __attribute__((ext_vector_type(8)))  __bf16 v8bf;
typedef __attribute__((ext_vector_type(8)))  float  v8f;

typedef __attribute__((ext_vector_type(4))) unsigned int u32x4;
typedef __attribute__((ext_vector_type(4))) int          i32x4;
typedef __attribute__((ext_vector_type(8))) int          i32x8;

// Problem constants
#define N_IMG 128          // B*T
#define CCH   64           // channels (in == out)
#define HWPIX 4096         // 64*64

// ---------------------------------------------------------------------------
// Kernel 1: f32 NCHW -> bf16 NHWC  (33,554,432 elements, exact grid)
// ---------------------------------------------------------------------------
__global__ __launch_bounds__(256) void pack_x_kernel(const float* __restrict__ x,
                                                     __bf16* __restrict__ xb) {
  int idx = blockIdx.x * 256 + threadIdx.x;
  int c  = idx & 63;
  int hw = (idx >> 6) & 4095;
  int n  = idx >> 18;
  float v = x[((size_t)(n * 64 + c)) * 4096 + hw];
  xb[idx] = (__bf16)v;
}

// ---------------------------------------------------------------------------
// Kernel 2: conv_w OIHW f32 -> bf16 [tap][co][ci]   (36,864 elements)
// ---------------------------------------------------------------------------
__global__ __launch_bounds__(256) void pack_w_kernel(const float* __restrict__ w,
                                                     __bf16* __restrict__ wp) {
  int idx = blockIdx.x * 256 + threadIdx.x;   // (tap, co, ci)
  int ci  = idx & 63;
  int co  = (idx >> 6) & 63;
  int tap = idx >> 12;                        // 0..8
  wp[idx] = (__bf16)w[(co * 64 + ci) * 9 + tap];
}

// ---------------------------------------------------------------------------
// TDM: DMA one 3x3-tap weight tile (64x64 bf16 = 8 KB, contiguous) into LDS.
// D# per CDNA5 ISA ch.8: group0 = {count/type, lds_addr, global_addr},
// group1 = {data_size, tensor dims/strides, tile dims}. Groups 2/3 zero
// (tile_dim3/4 = 0 -> unused). Tracked with TENSORcnt.
// ---------------------------------------------------------------------------
__device__ __forceinline__ void tdm_load_tap(const __bf16* gsrc, __bf16* ldst) {
  unsigned long long ga = (unsigned long long)reinterpret_cast<uintptr_t>(gsrc);
  unsigned lds = (unsigned)reinterpret_cast<uintptr_t>(ldst);  // LDS offset = low 32 bits

  u32x4 g0;
  g0[0] = 1u;                                            // count=1, user mode
  g0[1] = lds;                                           // lds_addr
  g0[2] = (unsigned)(ga & 0xffffffffu);                  // global_addr[31:0]
  g0[3] = (unsigned)((ga >> 32) & 0x01ffffffu)           // global_addr[56:32]
        | (2u << 30);                                    // type = 2 ("image")

  i32x8 g1;
  g1[0] = (int)(1u << 16);       // data_size = 1 (2-byte elements)
  g1[1] = (int)(4096u << 16);    // tensor_dim0[15:0] = 4096  (bits 63:48)
  g1[2] = (int)(1u << 16);       // tensor_dim1 = 1           (bits 111:80)
  g1[3] = (int)(4096u << 16);    // tile_dim0 = 4096          (bits 127:112)
  g1[4] = 1;                     // tile_dim1 = 1, tile_dim2 = 0
  g1[5] = 4096;                  // tensor_dim0_stride[31:0]
  g1[6] = 0;                     // stride hi / tensor_dim1_stride lo
  g1[7] = 0;

  i32x4 z4 = {};
#if defined(__clang_major__) && (__clang_major__ >= 23)
  i32x8 z8 = {};
  __builtin_amdgcn_tensor_load_to_lds(g0, g1, z4, z4, z8, 0);
#else
  __builtin_amdgcn_tensor_load_to_lds(g0, g1, z4, z4, 0);
#endif
}

// ---------------------------------------------------------------------------
// Kernel 3: implicit-GEMM 3x3 conv (bf16 WMMA, f32 accum) + fused LayerNorm.
// Block: 256 threads = 8 waves, one image, 128 pixels x all 64 co.
// Weights double-buffered in LDS via TDM tensor_load_to_lds (wave 0 issues,
// s_wait_tensorcnt keeps one transfer in flight to overlap DMA with WMMA).
// ---------------------------------------------------------------------------
__global__ __launch_bounds__(256) void conv_ln_kernel(
    const __bf16* __restrict__ xb,   // [128][64][64][64] bf16 NHWC
    const __bf16* __restrict__ wp,   // [9][64][64] bf16 (tap, co, ci)
    const float*  __restrict__ bias,
    const float*  __restrict__ gamma,
    const float*  __restrict__ beta,
    __bf16*       __restrict__ yn)   // [128][64][64][64] bf16 NHWC
{
  __shared__ __bf16 wlds[2][64 * 64];     // double-buffered tap weights (16 KB)
  __shared__ float  ytile[128][65];       // conv result tile (~33 KB, padded)
  __shared__ float  bias_l[64], gam_l[64], bet_l[64];

  const int tid = threadIdx.x;
  const int n       = blockIdx.x >> 5;          // image 0..127
  const int pixbase = (blockIdx.x & 31) * 128;  // 128 pixels per block

  if (tid < 64) {
    bias_l[tid] = bias[tid];
    gam_l[tid]  = gamma[tid];
    bet_l[tid]  = beta[tid];
  }

  const int wv   = tid >> 5;        // wave 0..7
  const int lane = tid & 31;
  const int m    = lane & 15;       // A-fragment row (pixel within tile)
  const int half = lane >> 4;       // 0: K lo half, 1: K hi half
  const int col  = lane & 15;       // B/D column (co within 16-tile)

  const int pix = pixbase + wv * 16 + m;  // 0..4095 within image
  const int h   = pix >> 6;
  const int w   = pix & 63;

  const __bf16* xn = xb + (size_t)n * (HWPIX * 64);

  v8f acc[4] = {};   // 4 co-tiles of 16x16 f32

  if (wv == 0) tdm_load_tap(wp, &wlds[0][0]);   // prime tap 0

  for (int tap = 0; tap < 9; ++tap) {
    const int cur = tap & 1;
    if (wv == 0) {
      if (tap + 1 < 9) {
        tdm_load_tap(wp + (tap + 1) * 4096, &wlds[cur ^ 1][0]);  // prefetch next
        __builtin_amdgcn_s_wait_tensorcnt(1);   // current buf done, next in flight
      } else {
        __builtin_amdgcn_s_wait_tensorcnt(0);   // last tap: drain
      }
    }
    __syncthreads();   // publish wlds[cur] to all waves

    const int dh = tap / 3 - 1;
    const int dw = tap - (tap / 3) * 3 - 1;
    const int h2 = h + dh;
    const int w2 = w + dw;
    const bool valid = ((unsigned)h2 < 64u) && ((unsigned)w2 < 64u);

#pragma unroll
    for (int kc = 0; kc < 2; ++kc) {   // two 32-wide ci chunks
      // ---- A fragment: this lane's pixel, 16 contiguous K-values (k = 16*half + s) ----
      v16bf av = {};
      if (valid) {
        const __bf16* p = xn + (size_t)((h2 << 6) + w2) * 64 + kc * 32 + half * 16;
        v8bf lo = *(const v8bf*)p;          // K: base..base+7
        v8bf hi = *(const v8bf*)(p + 8);    // K: base+8..base+15
#pragma unroll
        for (int j = 0; j < 8; ++j) { av[j] = lo[j]; av[8 + j] = hi[j]; }
      }

      const int kb = kc * 32 + half * 16;   // B fragment K base for this lane
#pragma unroll
      for (int ct = 0; ct < 4; ++ct) {      // 4 output-channel tiles
        v16bf bv = *(const v16bf*)(&wlds[cur][(ct * 16 + col) * 64 + kb]);
        acc[ct] = __builtin_amdgcn_wmma_f32_16x16x32_bf16(
            /*neg_a=*/false, av, /*neg_b=*/false, bv,
            /*c_mod=*/(short)0, acc[ct], /*reuse_a=*/false, /*reuse_b=*/false);
      }
    }
    __syncthreads();   // all reads of wlds[cur] done before it is re-targeted
  }

  // ---- scatter accumulators to LDS: D element (r) is pixel (half*8 + r) ----
#pragma unroll
  for (int ct = 0; ct < 4; ++ct) {
#pragma unroll
    for (int r = 0; r < 8; ++r) {
      ytile[wv * 16 + half * 8 + r][ct * 16 + col] = acc[ct][r];
    }
  }
  __syncthreads();

  // ---- fused LayerNorm over channels, one thread per pixel ----
  if (tid < 128) {
    const int p    = tid;
    const int gpix = pixbase + p;

    float s = 0.f, ss = 0.f;
#pragma unroll
    for (int c = 0; c < 64; ++c) {
      float y = ytile[p][c] + bias_l[c];
      s  += y;
      ss += y * y;
    }
    const float mean = s * (1.0f / 64.0f);
    const float var  = ss * (1.0f / 64.0f) - mean * mean;
    const float rstd = rsqrtf(var + 1e-5f);

    __bf16* dst = yn + ((size_t)n * HWPIX + gpix) * 64;
#pragma unroll
    for (int c0 = 0; c0 < 64; c0 += 8) {
      v8bf pack;
#pragma unroll
      for (int j = 0; j < 8; ++j) {
        float y  = ytile[p][c0 + j] + bias_l[c0 + j];
        float z  = (y - mean) * rstd * gam_l[c0 + j] + bet_l[c0 + j];
        pack[j] = (__bf16)z;
      }
      *(v8bf*)(dst + c0) = pack;
    }
  }
}

// ---------------------------------------------------------------------------
// Kernel 4: LIF scan over T=16. One thread per (b, c, hw); hw fastest so the
// f32 NCHW spike stores are fully coalesced.
// ---------------------------------------------------------------------------
__global__ __launch_bounds__(256) void lif_kernel(const __bf16* __restrict__ yn,
                                                  float* __restrict__ out) {
  int idx = blockIdx.x * 256 + threadIdx.x;   // (b, c, hw), 2,097,152 total
  int hw = idx & 4095;
  int c  = (idx >> 12) & 63;
  int b  = idx >> 18;

  float v = 0.f;
#pragma unroll
  for (int t = 0; t < 16; ++t) {
    int nimg = b * 16 + t;
    float x = (float)yn[((size_t)nimg * HWPIX + hw) * 64 + c];
    v = 0.5f * (v + x);                       // v + (x - v)/tau, tau = 2
    float sp = (v > 1.0f) ? 1.0f : 0.0f;      // hard heaviside (forward)
    out[((size_t)(nimg * 64 + c)) * 4096 + hw] = sp;
    v *= (1.0f - sp);                         // hard reset
  }
}

// ---------------------------------------------------------------------------
// Launch: pack -> conv+LN -> LIF, all stream-ordered.
// Workspace: [0,73728) packed weights | bf16 NHWC input | bf16 NHWC yn
// Total ~128.1 MiB.
// ---------------------------------------------------------------------------
extern "C" void kernel_launch(void* const* d_in, const int* in_sizes, int n_in,
                              void* d_out, int out_size, void* d_ws, size_t ws_size,
                              hipStream_t stream) {
  const float* x     = (const float*)d_in[0];   // [8,16,64,64,64]
  const float* cw    = (const float*)d_in[1];   // [64,64,3,3]
  const float* cb    = (const float*)d_in[2];   // [64]
  const float* gamma = (const float*)d_in[3];   // [64]
  const float* beta  = (const float*)d_in[4];   // [64]
  float* out = (float*)d_out;                   // [8,16,64,64,64] f32 spikes

  char* ws = (char*)d_ws;
  __bf16* Wp = (__bf16*)ws;                                  // 73,728 B
  __bf16* Xb = (__bf16*)(ws + 73728);                        // 67,108,864 B
  __bf16* Yn = (__bf16*)(ws + 73728 + 67108864);             // 67,108,864 B

  pack_x_kernel <<<131072, 256, 0, stream>>>(x, Xb);
  pack_w_kernel <<<144,    256, 0, stream>>>(cw, Wp);
  conv_ln_kernel<<<4096,   256, 0, stream>>>(Xb, Wp, cb, gamma, beta, Yn);
  lif_kernel    <<<8192,   256, 0, stream>>>(Yn, out);
}
